// MyProteinFeatures_33535104647252
// MI455X (gfx1250) — matrix operand order repaired and checked by reading, not compiled, not run on hardware
//
#include <hip/hip_runtime.h>
#include <hip/hip_fp16.h>

#define NRES   2048
#define BATCH  4
#define TOPK   32
#define EDGE_IN 416
#define EF     128
#define KSTEPS 13   // 13*32 = 416

typedef __attribute__((ext_vector_type(16))) _Float16     v16h;
typedef __attribute__((ext_vector_type(8)))  float        v8f;
typedef __attribute__((ext_vector_type(8)))  unsigned int v8u;

// ---------------------------------------------------------------------------
// Kernel 1: per-residue atom table [b*N][5][3] with atoms {N, Ca, C, O, Cb}
// ---------------------------------------------------------------------------
__global__ void coords_kernel(const float* __restrict__ X, float* __restrict__ atoms5) {
    int idx = blockIdx.x * blockDim.x + threadIdx.x;
    if (idx >= BATCH * NRES) return;
    const float* p = X + (size_t)idx * 12;   // [4 atoms][3]
    float n0=p[0],n1=p[1],n2=p[2];
    float a0=p[3],a1=p[4],a2=p[5];
    float c0=p[6],c1=p[7],c2=p[8];
    float o0=p[9],o1=p[10],o2=p[11];
    float bx=a0-n0, by=a1-n1, bz=a2-n2;
    float cx=c0-a0, cy=c1-a1, cz=c2-a2;
    float ax=by*cz-bz*cy, ay=bz*cx-bx*cz, az=bx*cy-by*cx;
    float cb0 = -0.58273431f*ax + 0.56802827f*bx - 0.54067466f*cx + a0;
    float cb1 = -0.58273431f*ay + 0.56802827f*by - 0.54067466f*cy + a1;
    float cb2 = -0.58273431f*az + 0.56802827f*bz - 0.54067466f*cz + a2;
    float* q = atoms5 + (size_t)idx * 15;
    q[0]=n0;  q[1]=n1;  q[2]=n2;      // N
    q[3]=a0;  q[4]=a1;  q[5]=a2;      // Ca
    q[6]=c0;  q[7]=c1;  q[8]=c2;      // C
    q[9]=o0;  q[10]=o1; q[11]=o2;     // O
    q[12]=cb0;q[13]=cb1;q[14]=cb2;    // Cb
}

// ---------------------------------------------------------------------------
// Kernel 2: pack W_edge (416x128 f32) into f16 B-fragment order.
// B 32x16 f16 layout: lanes 0-15 -> N=lane, K={0..7,16..23};
//                     lanes 16-31 -> N=lane-16, K={8..15,24..31} (VGPR v -> pair 2*(v&3) + 16*(v>>2) + 8*half).
// Storage index: kk*2048 + nt*256 + lane*8 + v   (dwords)
// ---------------------------------------------------------------------------
__global__ void wpack_kernel(const float* __restrict__ W, unsigned int* __restrict__ wfrag) {
    int t = blockIdx.x * blockDim.x + threadIdx.x;
    if (t >= KSTEPS * 8 * 32 * 8) return;
    int v    = t & 7;
    int lane = (t >> 3) & 31;
    int nt   = (t >> 8) & 7;
    int kk   = t >> 11;
    int n    = nt * 16 + (lane & 15);
    int kl   = (v >> 2) * 16 + ((lane < 16) ? 0 : 8) + 2 * (v & 3);
    int k0   = kk * 32 + kl;
    unsigned short h0 = __builtin_bit_cast(unsigned short, (_Float16)W[(size_t)k0 * EF + n]);
    unsigned short h1 = __builtin_bit_cast(unsigned short, (_Float16)W[(size_t)(k0 + 1) * EF + n]);
    wfrag[t] = (unsigned)h0 | ((unsigned)h1 << 16);
}

// ---------------------------------------------------------------------------
// Kernel 3: top-32 nearest neighbors per row. One 256-thread block per (b,i).
// Key = (f32bits(D_adjust) << 32) | j  -> min key = min D, ties -> min index.
// ---------------------------------------------------------------------------
__global__ __launch_bounds__(256) void topk_kernel(
    const float* __restrict__ atoms5, const float* __restrict__ mask,
    int* __restrict__ eidx, float* __restrict__ dnbuf, int* __restrict__ outidx) {
    __shared__ unsigned long long sRedK[8];
    __shared__ float sRedF[8];
    const int row  = blockIdx.x;           // b*NRES + i
    const int b    = row >> 11;
    const int tid  = threadIdx.x;
    const int lane = tid & 31, wid = tid >> 5;
    const float cx = atoms5[row*15+3], cy = atoms5[row*15+4], cz = atoms5[row*15+5];
    const float mi = mask[row];

    float Dloc[8];
    float lmax = 0.0f;
    #pragma unroll
    for (int c = 0; c < 8; ++c) {
        int jj = tid + c * 256;
        const float* pj = atoms5 + (size_t)(b * NRES + jj) * 15;
        float dx = cx - pj[3], dy = cy - pj[4], dz = cz - pj[5];
        float m2 = mi * mask[b * NRES + jj];
        float D  = m2 * sqrtf(dx*dx + dy*dy + dz*dz + 1e-6f);
        Dloc[c] = D;
        lmax = fmaxf(lmax, D);
    }
    #pragma unroll
    for (int off = 16; off; off >>= 1) lmax = fmaxf(lmax, __shfl_xor(lmax, off, 32));
    if (lane == 0) sRedF[wid] = lmax;
    __syncthreads();
    float Dmax = sRedF[0];
    #pragma unroll
    for (int w = 1; w < 8; ++w) Dmax = fmaxf(Dmax, sRedF[w]);

    unsigned long long kloc[8];
    #pragma unroll
    for (int c = 0; c < 8; ++c) {
        int jj = tid + c * 256;
        float m2 = mi * mask[b * NRES + jj];
        float Dadj = Dloc[c] + (1.0f - m2) * Dmax;
        kloc[c] = ((unsigned long long)__float_as_uint(Dadj) << 32) | (unsigned)jj;
    }
    for (int it = 0; it < TOPK; ++it) {
        unsigned long long lmin = kloc[0];
        #pragma unroll
        for (int c = 1; c < 8; ++c) lmin = (kloc[c] < lmin) ? kloc[c] : lmin;
        #pragma unroll
        for (int off = 16; off; off >>= 1) {
            unsigned long long o = __shfl_xor(lmin, off, 32);
            lmin = (o < lmin) ? o : lmin;
        }
        if (lane == 0) sRedK[wid] = lmin;
        __syncthreads();
        unsigned long long w = sRedK[0];
        #pragma unroll
        for (int t = 1; t < 8; ++t) { unsigned long long o = sRedK[t]; w = (o < w) ? o : w; }
        #pragma unroll
        for (int c = 0; c < 8; ++c) if (kloc[c] == w) kloc[c] = ~0ULL;
        if (tid == 0) {
            int   jw = (int)(unsigned)(w & 0xFFFFFFFFu);
            float Dw = __uint_as_float((unsigned)(w >> 32));
            eidx  [row * TOPK + it] = jw;
            dnbuf [row * TOPK + it] = Dw;
            outidx[row * TOPK + it] = jw;
        }
        __syncthreads();
    }
}

// ---------------------------------------------------------------------------
// Kernel 4: fused feature -> WMMA GEMM (416->128, f16) -> LayerNorm -> store.
// One wave per 16-edge tile (all 16 edges share residue i). 8 waves/block.
// A fragment (16x32 f16): lane half owns K {0..7,16..23} or {8..15,24..31};
// per K-step only two of the 25 distances are touched: dist[2kk-1], dist[2kk].
// ---------------------------------------------------------------------------
__global__ __launch_bounds__(256) void edge_kernel(
    const float* __restrict__ atoms5, const float* __restrict__ dnb,
    const int* __restrict__ eidx, const unsigned int* __restrict__ wfrag,
    const float* __restrict__ gamma, const float* __restrict__ beta,
    float* __restrict__ out) {
    const int lane = threadIdx.x & 31;
    const int wid  = threadIdx.x >> 5;
    const int tile = blockIdx.x * 8 + wid;    // 16384 tiles of 16 edges
    const int row  = tile >> 1;               // b*NRES + i
    const int ksub = (tile & 1) * 16;
    const int m    = lane & 15;
    const int hf   = lane >> 4;               // half-wave id
    const int ho   = hf * 8;

    const int i = row & (NRES - 1);
    const int b = row >> 11;
    const int k = ksub + m;
    const int eoff = row * TOPK + k;
    const int j = eidx[eoff];
    const float dneb = dnb[eoff];
    const float dpos = (float)j - (float)i;

    const float* xi = atoms5 + (size_t)row * 15;
    const float* xj = atoms5 + (size_t)(b * NRES + j) * 15;
    float xiL[15], xjL[15];
    #pragma unroll
    for (int t = 0; t < 15; ++t) { xiL[t] = xi[t]; xjL[t] = xj[t]; }

    // pair table (A=residue i atom, B=neighbor j atom); atoms {0:N,1:Ca,2:C,3:O,4:Cb}
    constexpr int AI[24] = {0,2,3,4,1,1,1,1,0,0,0,4,4,3,0,2,3,4,2,3,4,2,3,2};
    constexpr int BI[24] = {0,2,3,4,0,2,3,4,2,3,4,2,3,2,1,1,1,1,0,0,0,4,4,3};
    float dist[25];
    dist[0] = dneb;
    #pragma unroll
    for (int r = 0; r < 24; ++r) {
        float dx = xiL[AI[r]*3+0] - xjL[BI[r]*3+0];
        float dy = xiL[AI[r]*3+1] - xjL[BI[r]*3+1];
        float dz = xiL[AI[r]*3+2] - xjL[BI[r]*3+2];
        dist[r+1] = sqrtf(dx*dx + dy*dy + dz*dz + 1e-6f);
    }

    const float FREQ[8] = {1.0f, 0.31622776601683794f, 0.1f, 0.031622776601683794f,
                           0.01f, 0.0031622776601683794f, 0.001f, 0.00031622776601683794f};

    v8f acc[8];
    #pragma unroll
    for (int nt = 0; nt < 8; ++nt) { v8f z = {}; acc[nt] = z; }

    #pragma unroll
    for (int kk = 0; kk < KSTEPS; ++kk) {
        v16h a;
        const float dA = dist[kk == 0 ? 0 : 2*kk - 1];
        const float dB = dist[kk == 0 ? 0 : 2*kk];
        #pragma unroll
        for (int v = 0; v < 4; ++v) {
            #pragma unroll
            for (int e = 0; e < 2; ++e) {
                float val;
                if (kk == 0) {
                    float ang = dpos * FREQ[2*v + e];
                    val = hf ? __sinf(ang) : __cosf(ang);
                } else {
                    float mu = 2.0f + 1.3333333333333333f * (float)(ho + 2*v + e);
                    float t  = (dA - mu) * 0.8f;
                    val = __expf(-t * t);
                }
                a[2*v + e] = (_Float16)val;
            }
        }
        #pragma unroll
        for (int v = 4; v < 8; ++v) {
            #pragma unroll
            for (int e = 0; e < 2; ++e) {
                float mu = 2.0f + 1.3333333333333333f * (float)(ho + 2*(v-4) + e);
                float t  = (dB - mu) * 0.8f;
                a[2*v + e] = (_Float16)__expf(-t * t);
            }
        }
        #pragma unroll
        for (int nt = 0; nt < 8; ++nt) {
            v8u wraw = *(const v8u*)(wfrag + (kk * 2048 + nt * 256 + lane * 8));
            v16h bf  = __builtin_bit_cast(v16h, wraw);
            acc[nt] = __builtin_amdgcn_wmma_f32_16x16x32_f16(
                false, a, false, bf, (short)0, acc[nt], false, false);
        }
    }

    // LayerNorm over 128 outputs per edge. D layout: lane holds col n=nt*16+m,
    // rows v+ho; each row lives entirely in one 16-lane half.
    float s1[8], s2[8];
    #pragma unroll
    for (int v = 0; v < 8; ++v) {
        float s = 0.0f, q = 0.0f;
        #pragma unroll
        for (int nt = 0; nt < 8; ++nt) { float x = acc[nt][v]; s += x; q += x * x; }
        #pragma unroll
        for (int off = 1; off < 16; off <<= 1) {
            s += __shfl_xor(s, off, 32);
            q += __shfl_xor(q, off, 32);
        }
        s1[v] = s; s2[v] = q;
    }

    float g[8], bb[8];
    #pragma unroll
    for (int nt = 0; nt < 8; ++nt) { g[nt] = gamma[nt*16 + m]; bb[nt] = beta[nt*16 + m]; }

    const int ebase = row * TOPK + ksub;
    #pragma unroll
    for (int v = 0; v < 8; ++v) {
        float mu   = s1[v] * (1.0f / 128.0f);
        float var  = s2[v] * (1.0f / 128.0f) - mu * mu;
        float rstd = rsqrtf(var + 1e-5f);
        float* op  = out + (size_t)(ebase + v + ho) * EF;
        #pragma unroll
        for (int nt = 0; nt < 8; ++nt) {
            op[nt*16 + m] = (acc[nt][v] - mu) * rstd * g[nt] + bb[nt];
        }
    }
}

// ---------------------------------------------------------------------------
extern "C" void kernel_launch(void* const* d_in, const int* in_sizes, int n_in,
                              void* d_out, int out_size, void* d_ws, size_t ws_size,
                              hipStream_t stream) {
    (void)in_sizes; (void)n_in; (void)out_size; (void)ws_size;
    const float* X     = (const float*)d_in[0];
    const float* mask  = (const float*)d_in[1];
    const float* W     = (const float*)d_in[2];
    const float* gamma = (const float*)d_in[3];
    const float* beta  = (const float*)d_in[4];

    float* outE   = (float*)d_out;
    int*   outIdx = (int*)d_out + (size_t)BATCH * NRES * TOPK * EF;

    float*        atoms5 = (float*)d_ws;                       // 122880 f32
    float*        dnb    = atoms5 + BATCH * NRES * 15;          // 262144 f32
    int*          eidx   = (int*)(dnb + BATCH * NRES * TOPK);   // 262144 i32
    unsigned int* wfrag  = (unsigned int*)(eidx + BATCH * NRES * TOPK); // 26624 u32

    coords_kernel<<<(BATCH * NRES + 255) / 256, 256, 0, stream>>>(X, atoms5);
    wpack_kernel<<<(KSTEPS * 8 * 32 * 8 + 255) / 256, 256, 0, stream>>>(W, wfrag);
    topk_kernel<<<BATCH * NRES, 256, 0, stream>>>(atoms5, mask, eidx, dnb, outIdx);
    edge_kernel<<<(BATCH * NRES * TOPK) / 128, 256, 0, stream>>>(
        atoms5, dnb, eidx, wfrag, gamma, beta, outE);
}
